// GraphMolActorCritic_thv1_14834817040728
// MI455X (gfx1250) — matrix operand order repaired and checked by reading, not compiled, not run on hardware
//
#include <hip/hip_runtime.h>
#include <hip/hip_bf16.h>

// ---------------- problem constants (from reference) ----------------
#define N_NODES 32768
#define E_EDGES 65536
#define B_GRAPHS 1024
#define NPG 32      // nodes per graph
#define EPG 64      // edges per graph
#define S_STEMS 8
#define J_BONDS 4
#define GD 64       // hidden dim D
#define F_IN 14
#define NOPS 105
#define A_ACT 845   // 1 + 8*105 + 4
#define OUTC 846    // value + masked logits

typedef _Float16 v16h __attribute__((ext_vector_type(16)));
typedef _Float16 v8h  __attribute__((ext_vector_type(8)));
typedef float    v8f  __attribute__((ext_vector_type(8)));

__device__ __forceinline__ float lrelu(float v) { return v > 0.0f ? v : 0.01f * v; }
__device__ __forceinline__ float sigm(float v)  { return 1.0f / (1.0f + expf(-v)); }

// ---------------------------------------------------------------------
// Kernel 1: edge MLP stage 1  h1[e,k] = leaky(edge_attr[e,:4] @ net1_w + b), f16
// ---------------------------------------------------------------------
__global__ __launch_bounds__(256) void k_prep(const float* __restrict__ ea,
                                              const float* __restrict__ w1,
                                              const float* __restrict__ b1,
                                              _Float16* __restrict__ h1) {
  int idx = blockIdx.x * 256 + threadIdx.x;   // E*64 items
  int e = idx >> 6, k = idx & 63;
  float acc = b1[k];
#pragma unroll
  for (int c = 0; c < 4; ++c) acc += ea[e * 4 + c] * w1[c * GD + k];
  h1[idx] = (_Float16)lrelu(acc);
}

// Kernel 2: convert net2_w (64 x 4096) to f16 once
__global__ __launch_bounds__(256) void k_cvt(const float* __restrict__ w,
                                             _Float16* __restrict__ o) {
  int i = blockIdx.x * 256 + threadIdx.x;
  o[i] = (_Float16)w[i];
}

// ---------------------------------------------------------------------
// Kernel 3: Wmat[e, d*64+f] = h1[e,:] @ net2_w[:, d*64+f] + net2_b  (f16 out)
// GEMM M=65536, K=64, N=4096 via v_wmma_f32_16x16x32_f16 (two K=32 steps).
// ---------------------------------------------------------------------
__global__ __launch_bounds__(256) void k_wmat(const _Float16* __restrict__ h1,
                                              const _Float16* __restrict__ w2,
                                              const float* __restrict__ b2,
                                              _Float16* __restrict__ wm) {
  __shared__ alignas(16) _Float16 aT[16 * 64];    // 16 edges x K64
  __shared__ alignas(16) _Float16 bT[64 * 128];   // K64 x N128
  __shared__ alignas(16) _Float16 dT[16 * 128];   // M16 x N128 result staging
  const int tid = threadIdx.x;
  const int lane = tid & 31, wid = tid >> 5;
  const int m0 = blockIdx.x * 16;       // edge-tile base
  const int n0blk = blockIdx.y * 128;   // column base

  { // stage A rows (coalesced, 512 dwords)
    unsigned int* d = (unsigned int*)aT;
    const unsigned int* s = (const unsigned int*)(h1 + (size_t)m0 * GD);
    for (int u = tid; u < 512; u += 256) d[u] = s[u];
  }
  { // stage B tile (64 rows x 64 dwords)
    unsigned int* d = (unsigned int*)bT;
    const unsigned int* s = (const unsigned int*)w2;   // row k = 2048 dwords
    for (int u = tid; u < 4096; u += 256) {
      int r = u >> 6, c = u & 63;
      d[u] = s[(size_t)r * 2048 + (n0blk >> 1) + c];
    }
  }
  __syncthreads();

  // Build fragments per CDNA5 16-bit layouts (ISA 7.12.2).
  const int mrow = lane & 15;
  const int kbaseA = (lane >> 4) * 8;    // A: lanes16-31 start at K=8
  const int koffB  = (lane >> 4) * 8;    // B: lanes16-31 start at K=8 (same order as A)
  const int ncol = lane & 15;

  // A fragment halves are K-contiguous: two b128 LDS loads + shuffle each.
  const _Float16* ar = aT + mrow * 64;
  v8h a0lo = *(const v8h*)(ar + kbaseA);
  v8h a0hi = *(const v8h*)(ar + kbaseA + 16);
  v8h a1lo = *(const v8h*)(ar + 32 + kbaseA);
  v8h a1hi = *(const v8h*)(ar + 32 + kbaseA + 16);
  v16h a0 = __builtin_shufflevector(a0lo, a0hi, 0,1,2,3,4,5,6,7,8,9,10,11,12,13,14,15);
  v16h a1 = __builtin_shufflevector(a1lo, a1hi, 0,1,2,3,4,5,6,7,8,9,10,11,12,13,14,15);

  const int nw = wid * 16;
  v16h b0, b1;
#pragma unroll
  for (int e = 0; e < 16; ++e) {
    b0[e] = bT[(koffB + (e & 7) + ((e >> 3) << 4)) * 128 + nw + ncol];
    b1[e] = bT[(32 + koffB + (e & 7) + ((e >> 3) << 4)) * 128 + nw + ncol];
  }

  v8f c = {0.f, 0.f, 0.f, 0.f, 0.f, 0.f, 0.f, 0.f};
  c = __builtin_amdgcn_wmma_f32_16x16x32_f16(false, a0, false, b0, (short)0, c, false, false);
  c = __builtin_amdgcn_wmma_f32_16x16x32_f16(false, a1, false, b1, (short)0, c, false, false);

  float bias = b2[n0blk + nw + ncol];    // bias is per output column
#pragma unroll
  for (int gi = 0; gi < 8; ++gi) {       // C/D layout: M = gi + 8*(lane>=16), N = lane&15
    int Mr = gi + ((lane >> 4) * 8);
    dT[Mr * 128 + nw + ncol] = (_Float16)(c[gi] + bias);
  }
  __syncthreads();

  { // coalesced store: 16 rows x 64 dwords
    const unsigned int* s = (const unsigned int*)dT;
    unsigned int* d = (unsigned int*)wm;
    for (int u = tid; u < 1024; u += 256) {
      int r = u >> 6, cc = u & 63;
      d[(size_t)(m0 + r) * 2048 + (n0blk >> 1) + cc] = s[u];
    }
  }
}

// ---------------------------------------------------------------------
// Kernel 4: one block per graph — lin0, 6x(NNConv msg+agg+root+GRU),
// stem / jbond heads, Set2Set readout, masking, output assembly.
// Node state lives in LDS; only Wmat (f16, HBM-bound) is streamed with
// b128 loads + wave32 shfl_xor partial-sum reduction.
// ---------------------------------------------------------------------
struct Params {
  const float *x, *edge_attr;
  const int *edge_index, *stem_atmidx, *jbond_atmidx, *action_mask;
  const float *lin0_w, *lin0_b, *root_w, *conv_b;
  const float *gru_wi, *gru_wh, *gru_bi, *gru_bh;
  const float *ns1_w, *ns1_b, *ns2_w, *ns2_b;
  const float *nj1_w, *nj1_b, *nj2_w, *nj2_b;
  const float *lstm_bi, *lstm_bh, *lin_out_w, *lin_out_b;
  const _Float16 *wmat;
  float *out;
};

__global__ __launch_bounds__(256) void k_graph(Params p) {
  __shared__ alignas(16) float sOut[NPG * GD];   // node state (out == h)
  __shared__ alignas(16) float sNew[NPG * GD];   // GRU h_new staging
  __shared__ alignas(16) float sM[NPG * GD];     // m values; reused for heads
  __shared__ alignas(16) float sMsg[EPG * GD];   // per-edge messages
  __shared__ float sDeg[NPG];
  __shared__ int   sSrc[EPG], sDst[EPG];
  __shared__ float sQ[GD], sR[GD], sEv[NPG], sRed[2], sJB[8];

  const int g = blockIdx.x;
  const int tid = threadIdx.x;
  const int lane = tid & 31, wid = tid >> 5;

  if (tid < EPG) {
    int e = g * EPG + tid;
    sSrc[tid] = p.edge_index[e] - g * NPG;
    sDst[tid] = p.edge_index[E_EDGES + e] - g * NPG;
  }
  __syncthreads();
  if (tid < NPG) {   // deterministic degree count
    int c = 0;
    for (int j = 0; j < EPG; ++j) c += (sDst[j] == tid);
    sDeg[tid] = (c > 0) ? (float)c : 1.0f;
  }
  // lin0: out = leaky(x @ lin0_w + b)
  for (int t = 0; t < 8; ++t) {
    int idx = tid + t * 256; int n = idx >> 6; int f = idx & 63;
    const float* xr = p.x + (size_t)(g * NPG + n) * F_IN;
    float acc = p.lin0_b[f];
#pragma unroll
    for (int c = 0; c < F_IN; ++c) acc += xr[c] * p.lin0_w[c * GD + f];
    sOut[idx] = lrelu(acc);
  }
  __syncthreads();

  const int fg = lane & 7;    // lane owns f = fg*8 .. fg*8+7
  const int dr = lane >> 3;   // d-row within a chunk of 4

  for (int it = 0; it < 6; ++it) {
    // edge messages: wave-per-edge, 8 passes.
    // b128 loads: chunk ch covers d = 4*ch + dr; lane loads 8 f16 weights.
    for (int pp = 0; pp < 8; ++pp) {
      int j = pp * 8 + wid;
      int srcL = sSrc[j];
      const uint4* wp4 = (const uint4*)(p.wmat + (size_t)(g * EPG + j) * 4096);
      if (pp < 7) {  // prefetch next pass's 8KB tile (global_prefetch_b8)
        const char* nxt = (const char*)(p.wmat + (size_t)(g * EPG + j + 8) * 4096);
        __builtin_prefetch(nxt + lane * 256, 0, 1);
      }
      float acc[8] = {0.f, 0.f, 0.f, 0.f, 0.f, 0.f, 0.f, 0.f};
#pragma unroll 4
      for (int ch = 0; ch < 16; ++ch) {
        uint4 w = wp4[ch * 32 + lane];               // 512B coalesced / wave
        float s = sOut[srcL * GD + ch * 4 + dr];
        union { uint4 v; _Float16 h[8]; } cv; cv.v = w;
#pragma unroll
        for (int q = 0; q < 8; ++q) acc[q] += s * (float)cv.h[q];
      }
      // fold the 4 d-row partials (lanes ^8, ^16 share the same f-group)
#pragma unroll
      for (int q = 0; q < 8; ++q) {
        acc[q] += __shfl_xor(acc[q], 8, 32);
        acc[q] += __shfl_xor(acc[q], 16, 32);
      }
      if (dr == 0) {
        float4* dstv = (float4*)&sMsg[j * GD + fg * 8];
        float4 lo = {acc[0], acc[1], acc[2], acc[3]};
        float4 hi = {acc[4], acc[5], acc[6], acc[7]};
        dstv[0] = lo; dstv[1] = hi;
      }
    }
    __syncthreads();
    // deterministic scatter-free aggregation + root GEMM + activation
    for (int t = 0; t < 8; ++t) {
      int idx = tid + t * 256; int n = idx >> 6; int f = idx & 63;
      float agg = 0.f;
      for (int j = 0; j < EPG; ++j) if (sDst[j] == n) agg += sMsg[j * GD + f];
      agg /= sDeg[n];
      float rt = p.conv_b[f];
      for (int k = 0; k < GD; ++k) rt += sOut[n * GD + k] * p.root_w[k * GD + f];
      sM[idx] = lrelu(rt + agg);
    }
    __syncthreads();
    // GRU cell (torch order r,z,n)
    for (int t = 0; t < 8; ++t) {
      int idx = tid + t * 256; int n = idx >> 6; int f = idx & 63;
      const float* m = &sM[n * GD]; const float* h = &sOut[n * GD];
      float xr = p.gru_bi[f], xz = p.gru_bi[GD + f], xn = p.gru_bi[2 * GD + f];
      float hr = p.gru_bh[f], hz = p.gru_bh[GD + f], hn = p.gru_bh[2 * GD + f];
      const float* wir = p.gru_wi + (size_t)f * GD;
      const float* wiz = p.gru_wi + (size_t)(GD + f) * GD;
      const float* win = p.gru_wi + (size_t)(2 * GD + f) * GD;
      const float* whr = p.gru_wh + (size_t)f * GD;
      const float* whz = p.gru_wh + (size_t)(GD + f) * GD;
      const float* whn = p.gru_wh + (size_t)(2 * GD + f) * GD;
      for (int k = 0; k < GD; ++k) {
        float mk = m[k], hk = h[k];
        xr += mk * wir[k]; xz += mk * wiz[k]; xn += mk * win[k];
        hr += hk * whr[k]; hz += hk * whz[k]; hn += hk * whn[k];
      }
      float r = sigm(xr + hr);
      float z = sigm(xz + hz);
      float nn = tanhf(xn + r * hn);
      sNew[idx] = (1.f - z) * nn + z * h[f];
    }
    __syncthreads();
    for (int t = 0; t < 8; ++t) { int idx = tid + t * 256; sOut[idx] = sNew[idx]; }
    __syncthreads();
  }

  // ---- stem hidden (sM rows 0..7) and jbond hidden (sM rows 8..15) ----
  for (int t = 0; t < 2; ++t) {
    int idx = tid + t * 256; int s = idx >> 6; int f = idx & 63;
    int loc = p.stem_atmidx[g * S_STEMS + s] - g * NPG;
    float acc = p.ns1_b[f];
    for (int k = 0; k < GD; ++k) acc += sOut[loc * GD + k] * p.ns1_w[k * GD + f];
    sM[s * GD + f] = lrelu(acc);
  }
  for (int t = 0; t < 2; ++t) {
    int idx = tid + t * 256; int r = idx >> 6; int f = idx & 63;   // r = jj*2 + side
    int loc = p.jbond_atmidx[g * (J_BONDS * 2) + r] - g * NPG;
    float acc = p.nj1_b[f];
    for (int k = 0; k < GD; ++k) acc += sOut[loc * GD + k] * p.nj1_w[k * GD + f];
    sM[(8 + r) * GD + f] = lrelu(acc);
  }
  __syncthreads();

  // stem logits -> output cols [2 .. 841], mask cols [1 .. 840]
  for (int idx = tid; idx < S_STEMS * NOPS; idx += 256) {
    int s = idx / NOPS, o = idx % NOPS;
    float acc = p.ns2_b[o];
    for (int k = 0; k < GD; ++k) acc += sM[s * GD + k] * p.ns2_w[k * NOPS + o];
    int mcol = 1 + s * NOPS + o;
    float v = (p.action_mask[(size_t)g * A_ACT + mcol] == 1) ? acc : -20.0f;
    p.out[(size_t)g * OUTC + 1 + mcol] = v;
  }
  // jbond scalars
  if (tid < 8) {
    float acc = p.nj2_b[0];
    for (int k = 0; k < GD; ++k) acc += sM[(8 + tid) * GD + k] * p.nj2_w[k];
    sJB[tid] = acc;
  }
  // Set2Set q: qstar=0 so gates = biases only (lstm_wi term vanishes)
  if (tid < GD) {
    float gi_ = p.lstm_bi[tid] + p.lstm_bh[tid];
    float gc  = p.lstm_bi[2 * GD + tid] + p.lstm_bh[2 * GD + tid];
    float go  = p.lstm_bi[3 * GD + tid] + p.lstm_bh[3 * GD + tid];
    float cc = sigm(gi_) * tanhf(gc);
    sQ[tid] = sigm(go) * tanhf(cc);
  }
  __syncthreads();
  // break logits -> output cols [842..845], mask cols [841..844]
  if (tid < J_BONDS) {
    float v = 0.5f * (sJB[2 * tid] + sJB[2 * tid + 1]);
    int mcol = 1 + S_STEMS * NOPS + tid;
    float ov = (p.action_mask[(size_t)g * A_ACT + mcol] == 1) ? v : -20.0f;
    p.out[(size_t)g * OUTC + 1 + mcol] = ov;
  }
  // attention energies
  if (tid < NPG) {
    float acc = 0.f;
    for (int k = 0; k < GD; ++k) acc += sOut[tid * GD + k] * sQ[k];
    sEv[tid] = acc;
  }
  __syncthreads();
  if (tid == 0) {
    float mx = sEv[0];
    for (int n = 1; n < NPG; ++n) mx = fmaxf(mx, sEv[n]);
    float sm = 0.f;
    for (int n = 0; n < NPG; ++n) sm += expf(sEv[n] - mx);
    sRed[0] = mx; sRed[1] = sm;
  }
  __syncthreads();
  if (tid < GD) {   // attention-weighted readout r_read
    float mx = sRed[0], inv = 1.0f / sRed[1];
    float acc = 0.f;
    for (int n = 0; n < NPG; ++n) acc += expf(sEv[n] - mx) * sOut[n * GD + tid];
    sR[tid] = acc * inv;
  }
  __syncthreads();
  if (tid < 2) {    // scalar head on qstar = [q, r_read]
    float acc = p.lin_out_b[tid];
    for (int i = 0; i < GD; ++i) acc += sQ[i] * p.lin_out_w[i * 2 + tid];
    for (int i = 0; i < GD; ++i) acc += sR[i] * p.lin_out_w[(GD + i) * 2 + tid];
    if (tid == 0) {
      p.out[(size_t)g * OUTC + 0] = acc;   // value head
    } else {
      float ov = (p.action_mask[(size_t)g * A_ACT + 0] == 1) ? acc : -20.0f;
      p.out[(size_t)g * OUTC + 1] = ov;    // stop logit (mask col 0)
    }
  }
}

// ---------------------------------------------------------------------
extern "C" void kernel_launch(void* const* d_in, const int* in_sizes, int n_in,
                              void* d_out, int out_size, void* d_ws, size_t ws_size,
                              hipStream_t stream) {
  (void)in_sizes; (void)n_in; (void)out_size; (void)ws_size;
  const float* x           = (const float*)d_in[0];
  const float* edge_attr   = (const float*)d_in[1];
  const int*   edge_index  = (const int*)d_in[2];
  // d_in[3] = batch (implicit in per-graph blocking)
  const int*   stem_atmidx = (const int*)d_in[4];
  const int*   jbond_atmidx= (const int*)d_in[5];
  const int*   action_mask = (const int*)d_in[6];
  const float* lin0_w = (const float*)d_in[7];
  const float* lin0_b = (const float*)d_in[8];
  const float* net1_w = (const float*)d_in[9];
  const float* net1_b = (const float*)d_in[10];
  const float* net2_w = (const float*)d_in[11];
  const float* net2_b = (const float*)d_in[12];
  const float* root_w = (const float*)d_in[13];
  const float* conv_b = (const float*)d_in[14];
  const float* gru_wi = (const float*)d_in[15];
  const float* gru_wh = (const float*)d_in[16];
  const float* gru_bi = (const float*)d_in[17];
  const float* gru_bh = (const float*)d_in[18];
  const float* ns1_w  = (const float*)d_in[19];
  const float* ns1_b  = (const float*)d_in[20];
  const float* ns2_w  = (const float*)d_in[21];
  const float* ns2_b  = (const float*)d_in[22];
  const float* nj1_w  = (const float*)d_in[23];
  const float* nj1_b  = (const float*)d_in[24];
  const float* nj2_w  = (const float*)d_in[25];
  const float* nj2_b  = (const float*)d_in[26];
  // d_in[27] = lstm_wi (dead: qstar starts at zero)
  const float* lstm_bi   = (const float*)d_in[28];
  const float* lstm_bh   = (const float*)d_in[29];
  const float* lin_out_w = (const float*)d_in[30];
  const float* lin_out_b = (const float*)d_in[31];

  // workspace: Wmat f16 (512 MB) | h1 f16 (8 MB) | net2_w f16 (0.5 MB)
  char* ws = (char*)d_ws;
  _Float16* Wmat = (_Float16*)ws;
  _Float16* h1f  = (_Float16*)(ws + (size_t)E_EDGES * 4096 * 2);
  _Float16* w2f  = (_Float16*)(ws + (size_t)E_EDGES * 4096 * 2 + (size_t)E_EDGES * GD * 2);

  k_prep<<<(E_EDGES * GD) / 256, 256, 0, stream>>>(edge_attr, net1_w, net1_b, h1f);
  k_cvt<<<(GD * 4096) / 256, 256, 0, stream>>>(net2_w, w2f);
  dim3 gw(E_EDGES / 16, 4096 / 128);
  k_wmat<<<gw, 256, 0, stream>>>(h1f, w2f, net2_b, Wmat);

  Params p;
  p.x = x; p.edge_attr = edge_attr; p.edge_index = edge_index;
  p.stem_atmidx = stem_atmidx; p.jbond_atmidx = jbond_atmidx; p.action_mask = action_mask;
  p.lin0_w = lin0_w; p.lin0_b = lin0_b; p.root_w = root_w; p.conv_b = conv_b;
  p.gru_wi = gru_wi; p.gru_wh = gru_wh; p.gru_bi = gru_bi; p.gru_bh = gru_bh;
  p.ns1_w = ns1_w; p.ns1_b = ns1_b; p.ns2_w = ns2_w; p.ns2_b = ns2_b;
  p.nj1_w = nj1_w; p.nj1_b = nj1_b; p.nj2_w = nj2_w; p.nj2_b = nj2_b;
  p.lstm_bi = lstm_bi; p.lstm_bh = lstm_bh; p.lin_out_w = lin_out_w; p.lin_out_b = lin_out_b;
  p.wmat = Wmat; p.out = (float*)d_out;
  k_graph<<<B_GRAPHS, 256, 0, stream>>>(p);
}
